// SelCondPredictor_35527969472864
// MI455X (gfx1250) — matrix operand order repaired and checked by reading, not compiled, not run on hardware
//
#include <hip/hip_runtime.h>
#include <math.h>

// ---------------------------------------------------------------------------
// SelCondPredictor for MI455X (gfx1250, wave32, WMMA).
//
//  * All matmuls -> v_wmma_f32_16x16x32_f16 (f32 staged to f16 in LDS, f32
//    accumulate): K=32 per instruction vs K=4 on the f32 WMMA path.
//  * GEMM block tile 64x128 (8 waves, 2x2 16x16 tiles per wave): 4 WMMA per
//    wave per staging round with A/B fragment reuse; fragments are 2x
//    ds_load_b128 per lane (A staged [m][k], B staged [n][k], 80B rows).
//  * Interior blocks use uniform float4 staging; only edge tiles take the
//    guarded scalar path.
//  * BiLSTM: x@Wih^T hoisted into one parallel GEMM over all timesteps;
//    recurrence keeps Whh (512x128 f16) resident in the 320KB LDS and
//    issues 16 WMMA per step on an LDS-resident h tile.
// ---------------------------------------------------------------------------

#define BB   128   // batch
#define XDIM 256   // question length
#define CDIM 64    // number of columns
#define NW   300   // word embedding dim
#define NHD  256   // hidden (bi) dim
#define HD   128   // per-direction hidden
#define NEGV (-100.0f)

typedef __attribute__((ext_vector_type(16))) _Float16 v16h;
typedef __attribute__((ext_vector_type(8)))  _Float16 v8h;
typedef __attribute__((ext_vector_type(4)))  _Float16 v4h;
typedef __attribute__((ext_vector_type(8)))  float    v8f;

// --- param leaf indexing (JAX pytree flatten: dict keys sorted; each linear
// has leaves {W, b}; each lstm is [layer0, layer1] each {bw, fw} each
// {Whh, Wih, bhh, bih}). Leaf index is relative to d_in[2].
enum {
  L_agg_num_att = 0, L_agg_num_out_0 = 2, L_agg_num_out_1 = 4, L_agg_num_out_K = 6, L_agg_num_out_col = 8,
  L_agg_op_att = 10, L_agg_op_out_0 = 12, L_agg_op_out_1 = 14, L_agg_op_out_K = 16, L_agg_op_out_col = 18,
  L_gby_att = 20, L_gby_num_h = 22, L_gby_num_l = 24, L_gby_num_out = 26,
  L_gby_out = 28, L_gby_out_K = 30, L_gby_out_col = 32,
  L_lstm_col = 34, L_lstm_x = 50,
  L_ody_agg_att = 66, L_ody_agg_out_0 = 68, L_ody_agg_out_1 = 70, L_ody_agg_out_K = 72, L_ody_agg_out_col = 74,
  L_ody_att = 76, L_ody_num_h = 78, L_ody_num_l = 80, L_ody_num_out = 82,
  L_ody_out = 84, L_ody_out_K = 86, L_ody_out_col = 88,
  L_ody_par_h = 90, L_ody_par_l = 92, L_ody_par_out = 94,
  L_sel_att = 96, L_sel_num_h = 98, L_sel_num_l = 100, L_sel_num_out = 102,
  L_sel_out = 104, L_sel_out_K = 106, L_sel_out_col = 108,
  N_PARAM_LEAVES = 110
};

// Build a 16-half fragment from two 16B LDS runs.
__device__ __forceinline__ v16h frag16(const _Float16* p0, const _Float16* p1) {
  v8h lo = *(const v8h*)p0;
  v8h hi = *(const v8h*)p1;
  return __builtin_shufflevector(lo, hi, 0, 1, 2, 3, 4, 5, 6, 7,
                                 8, 9, 10, 11, 12, 13, 14, 15);
}

// ===========================================================================
// Generic WMMA GEMM:  C[M,N] = act( A[M,K] @ op(B)[K,N] + bias + accum*C )
//   transB=1: B is [N,K] row-major (weights);  transB=0: B is [K,N].
//   Batched via blockIdx.z. Block: 256 thr = 8 waves; block tile 64x128;
//   wave computes 2x2 16x16 tiles (wm = wave&1 selects M pair, wn = wave>>1
//   selects N pair).
// ===========================================================================
__global__ __launch_bounds__(256) void gemm_wmma(
    const float* __restrict__ A, long long ldA, long long sA,
    const float* __restrict__ Bm, long long ldB, long long sB, int transB,
    const float* __restrict__ bias,
    float* __restrict__ Cm, long long ldC, long long sC,
    int M, int N, int Kd, int act, int accum)
{
  __shared__ __align__(16) _Float16 As[64][40];   // [m][k], 80B rows
  __shared__ __align__(16) _Float16 Bs[128][40];  // [n][k], 80B rows
  const int tid  = threadIdx.x;
  const int wave = tid >> 5, lane = tid & 31;
  const int wm = wave & 1, wn = wave >> 1;
  const int mblk = blockIdx.y * 64, nblk = blockIdx.x * 128;
  const float* Ab = A  + (size_t)blockIdx.z * sA;
  const float* Bb = Bm + (size_t)blockIdx.z * sB;
  float*       Cb = Cm + (size_t)blockIdx.z * sC;
  const int row = lane & 15, half = lane >> 4;
  const bool interiorMN = (mblk + 64 <= M) && (nblk + 128 <= N);

  v8f acc[2][2];
  {
    v8f vz = {0.f, 0.f, 0.f, 0.f, 0.f, 0.f, 0.f, 0.f};
    acc[0][0] = vz; acc[0][1] = vz; acc[1][0] = vz; acc[1][1] = vz;
  }

  for (int k0 = 0; k0 < Kd; k0 += 32) {
    const bool interior = interiorMN && (k0 + 32 <= Kd);
    if (interior) {
      // ---- fast path: uniform float4 (b128) staging, no divergence ----
#pragma unroll
      for (int i = 0; i < 2; ++i) {           // A: 64x32 = 512 float4
        int e = tid + i * 256;
        int r = e >> 3, c = (e & 7) * 4;
        float4 va = *(const float4*)(Ab + (size_t)(mblk + r) * ldA + k0 + c);
        v4h hv; hv[0] = (_Float16)va.x; hv[1] = (_Float16)va.y;
                hv[2] = (_Float16)va.z; hv[3] = (_Float16)va.w;
        *(v4h*)&As[r][c] = hv;
      }
      if (transB) {
#pragma unroll
        for (int i = 0; i < 4; ++i) {         // B: 128x32 = 1024 float4
          int e = tid + i * 256;
          int n = e >> 3, c = (e & 7) * 4;
          float4 vb = *(const float4*)(Bb + (size_t)(nblk + n) * ldB + k0 + c);
          v4h hv; hv[0] = (_Float16)vb.x; hv[1] = (_Float16)vb.y;
                  hv[2] = (_Float16)vb.z; hv[3] = (_Float16)vb.w;
          *(v4h*)&Bs[n][c] = hv;
        }
      } else {
#pragma unroll
        for (int i = 0; i < 4; ++i) {         // B[k][n]: 32 k-rows x 32 float4
          int e = tid + i * 256;
          int kk = e >> 5, n4 = (e & 31) * 4;
          float4 vb = *(const float4*)(Bb + (size_t)(k0 + kk) * ldB + nblk + n4);
          Bs[n4 + 0][kk] = (_Float16)vb.x;
          Bs[n4 + 1][kk] = (_Float16)vb.y;
          Bs[n4 + 2][kk] = (_Float16)vb.z;
          Bs[n4 + 3][kk] = (_Float16)vb.w;
        }
      }
      if (k0 + 32 < Kd)  // speculative prefetch of next A chunk
        __builtin_prefetch(Ab + (size_t)(mblk + (tid >> 3)) * ldA + k0 + 32, 0, 1);
    } else {
      // ---- edge path: guarded scalar staging ----
#pragma unroll
      for (int i = 0; i < 8; ++i) {           // A: 64x32
        int e = tid + i * 256;
        int r = e >> 5, c = e & 31;
        int m = mblk + r, k = k0 + c;
        As[r][c] = (m < M && k < Kd) ? (_Float16)Ab[(size_t)m * ldA + k] : (_Float16)0.f;
      }
#pragma unroll
      for (int i = 0; i < 16; ++i) {          // B: 128x32
        int e = tid + i * 256;
        int nn = e >> 5, kk = e & 31;
        int n = nblk + nn, k = k0 + kk;
        float v = 0.f;
        if (n < N && k < Kd)
          v = transB ? Bb[(size_t)n * ldB + k] : Bb[(size_t)k * ldB + n];
        Bs[nn][kk] = (_Float16)v;
      }
    }
    __syncthreads();

    // Fragments (2x b128 each), reused across the 2x2 tile block.
    v16h afr[2], bfr[2];
#pragma unroll
    for (int im = 0; im < 2; ++im) {
      const _Float16* arow = &As[(wm * 2 + im) * 16 + row][0];
      afr[im] = frag16(arow + 16 * half, arow + 16 * half + 8);
    }
#pragma unroll
    for (int in_ = 0; in_ < 2; ++in_) {
      const _Float16* brow = &Bs[(wn * 2 + in_) * 16 + row][0];
      bfr[in_] = frag16(brow + 16 * half, brow + 16 * half + 8);
    }
#pragma unroll
    for (int im = 0; im < 2; ++im)
#pragma unroll
      for (int in_ = 0; in_ < 2; ++in_)
        acc[im][in_] = __builtin_amdgcn_wmma_f32_16x16x32_f16(
            false, afr[im], false, bfr[in_], (short)0, acc[im][in_], false, false);
    __syncthreads();
  }

  // Epilogue: lane holds (row = i + 8*half, col = lane&15) per the C layout.
#pragma unroll
  for (int im = 0; im < 2; ++im) {
#pragma unroll
    for (int in_ = 0; in_ < 2; ++in_) {
      int n = nblk + (wn * 2 + in_) * 16 + row;
      if (interiorMN) {
#pragma unroll
        for (int i = 0; i < 8; ++i) {
          int m = mblk + (wm * 2 + im) * 16 + i + 8 * half;
          float v = acc[im][in_][i];
          if (bias)  v += bias[n];
          if (accum) v += Cb[(size_t)m * ldC + n];
          if (act == 1) v = tanhf(v);
          Cb[(size_t)m * ldC + n] = v;
        }
      } else {
#pragma unroll
        for (int i = 0; i < 8; ++i) {
          int m = mblk + (wm * 2 + im) * 16 + i + 8 * half;
          if (m < M && n < N) {
            float v = acc[im][in_][i];
            if (bias)  v += bias[n];
            if (accum) v += Cb[(size_t)m * ldC + n];
            if (act == 1) v = tanhf(v);
            Cb[(size_t)m * ldC + n] = v;
          }
        }
      }
    }
  }
}

// ===========================================================================
// LSTM recurrence: one workgroup per 16 batch rows; Whh resident in LDS as
// f16 [512][136]; h tile in LDS; gates = Gx[t] + h @ Whh^T via 16 WMMA/step.
// PyTorch gate order i,f,g,o.
// ===========================================================================
#define WL_STRIDE 136
#define LSTM_SHMEM ((512 * WL_STRIDE + 16 * WL_STRIDE) * 2 + 16 * 512 * 4)

__global__ __launch_bounds__(256) void lstm_recur(
    const float* __restrict__ Gx,   // [B*T, 512] input gates (bias included)
    const float* __restrict__ Whh,  // [512, 128] f32
    const int*   __restrict__ lens,
    float*       __restrict__ Hout, // [B, T, 128]; zeroed at t >= len
    int T)
{
  extern __shared__ char smem[];
  _Float16* wl = (_Float16*)smem;                               // [512][136]
  _Float16* hl = wl + 512 * WL_STRIDE;                          // [16][136]
  float*    gl = (float*)(smem + (512 * WL_STRIDE + 16 * WL_STRIDE) * 2); // [16][512]

  const int tid  = threadIdx.x;
  const int wave = tid >> 5, lane = tid & 31;
  const int row  = lane & 15, half = lane >> 4;
  const int b0   = blockIdx.x * 16;

  // Load Whh into LDS once (reused across all T steps): float4 in, v4h out.
  for (int e = tid; e < 512 * 32; e += 256) {
    int n = e >> 5, c = (e & 31) * 4;
    float4 v = *(const float4*)(Whh + (size_t)n * 128 + c);
    v4h hv; hv[0] = (_Float16)v.x; hv[1] = (_Float16)v.y;
            hv[2] = (_Float16)v.z; hv[3] = (_Float16)v.w;
    *(v4h*)&wl[n * WL_STRIDE + c] = hv;
  }
  for (int e = tid; e < 16 * WL_STRIDE; e += 256) hl[e] = (_Float16)0.f;

  const int r  = tid >> 4;          // batch row within tile (0..15)
  const int j0 = (tid & 15) * 8;    // hidden slice start (8 per thread)
  float creg[8];
#pragma unroll
  for (int q = 0; q < 8; ++q) creg[q] = 0.f;
  const int mylen = lens[b0 + r];
  __syncthreads();

  for (int t = 0; t < T; ++t) {
    v8f vzero = {0.f, 0.f, 0.f, 0.f, 0.f, 0.f, 0.f, 0.f};
    v8f acc[4];
#pragma unroll
    for (int ntl = 0; ntl < 4; ++ntl) acc[ntl] = vzero;

#pragma unroll
    for (int kc = 0; kc < 4; ++kc) {
      const _Float16* arow = hl + row * WL_STRIDE + kc * 32;
      v16h af = frag16(arow + 16 * half, arow + 16 * half + 8);
#pragma unroll
      for (int ntl = 0; ntl < 4; ++ntl) {
        int nb = wave * 64 + ntl * 16;
        const _Float16* brow = wl + (size_t)(nb + row) * WL_STRIDE + kc * 32 + 16 * half;
        v16h bf = frag16(brow, brow + 8);
        acc[ntl] = __builtin_amdgcn_wmma_f32_16x16x32_f16(false, af, false, bf,
                                                          (short)0, acc[ntl], false, false);
      }
    }

    // WMMA result -> gl
#pragma unroll
    for (int ntl = 0; ntl < 4; ++ntl) {
#pragma unroll
      for (int i = 0; i < 8; ++i) {
        int rr  = i + 8 * half;
        int col = wave * 64 + ntl * 16 + row;
        gl[rr * 512 + col] = acc[ntl][i];
      }
    }
    const float* gxrow = Gx + ((size_t)(b0 + r) * T + t) * 512;
    if (t + 1 < T)
      __builtin_prefetch(gxrow + 512 + j0, 0, 1);
    __syncthreads();

    // gates = gl + Gx (coalesced float4), nonlinearity, state update
    float gv[4][8];
#pragma unroll
    for (int g = 0; g < 4; ++g) {
      float4 l0 = *(const float4*)&gl[r * 512 + g * 128 + j0];
      float4 l1 = *(const float4*)&gl[r * 512 + g * 128 + j0 + 4];
      float4 x0 = *(const float4*)&gxrow[g * 128 + j0];
      float4 x1 = *(const float4*)&gxrow[g * 128 + j0 + 4];
      gv[g][0] = l0.x + x0.x; gv[g][1] = l0.y + x0.y;
      gv[g][2] = l0.z + x0.z; gv[g][3] = l0.w + x0.w;
      gv[g][4] = l1.x + x1.x; gv[g][5] = l1.y + x1.y;
      gv[g][6] = l1.z + x1.z; gv[g][7] = l1.w + x1.w;
    }
    float hq[8];
#pragma unroll
    for (int q = 0; q < 8; ++q) {
      float si = 1.f / (1.f + __expf(-gv[0][q]));
      float sf = 1.f / (1.f + __expf(-gv[1][q]));
      float so = 1.f / (1.f + __expf(-gv[3][q]));
      float c  = sf * creg[q] + si * tanhf(gv[2][q]);
      creg[q] = c;
      hq[q] = so * tanhf(c);
    }
    // h -> global (masked) and LDS (b128 stores)
    {
      float* hout = Hout + ((size_t)(b0 + r) * T + t) * HD + j0;
      float m = (t < mylen) ? 1.f : 0.f;
      float4 o0, o1;
      o0.x = hq[0] * m; o0.y = hq[1] * m; o0.z = hq[2] * m; o0.w = hq[3] * m;
      o1.x = hq[4] * m; o1.y = hq[5] * m; o1.z = hq[6] * m; o1.w = hq[7] * m;
      *(float4*)hout = o0;
      *(float4*)(hout + 4) = o1;
      v8h hh;
#pragma unroll
      for (int q = 0; q < 8; ++q) hh[q] = (_Float16)hq[q];
      *(v8h*)&hl[r * WL_STRIDE + j0] = hh;
    }
    __syncthreads();
  }
}

// ===========================================================================
// Small helper kernels
// ===========================================================================
__global__ __launch_bounds__(512) void bias_sum(float* o, const float* a, const float* b) {
  int i = threadIdx.x;
  o[i] = a[i] + b[i];
}

// reversed (packed-sequence) gather: dst[b,t,:] = valid(b,t) * src[b, len-1-t, :]
// vectorized over float4 (D % 4 == 0 for D in {300, 256}).
__global__ __launch_bounds__(256) void gather_rev(float* dst, const float* src,
                                                  const int* lens, int T, int D) {
  int D4 = D >> 2;
  size_t i = (size_t)blockIdx.x * 256 + threadIdx.x;
  size_t total = (size_t)BB * T * D4;
  if (i >= total) return;
  int d = (int)(i % D4);
  size_t bt = i / D4;
  int t = (int)(bt % T);
  int b = (int)(bt / T);
  int len = lens[b];
  int rt = len - 1 - t;
  rt = rt < 0 ? 0 : (rt > T - 1 ? T - 1 : rt);
  float4 v = {0.f, 0.f, 0.f, 0.f};
  if (t < len) v = *(const float4*)(src + ((size_t)b * T + rt) * D + d * 4);
  *(float4*)(dst + i * 4) = v;
}

// out[b,t,0:128] = hfw ; out[b,t,128:256] = valid * hbw[b, len-1-t, :]
__global__ __launch_bounds__(256) void concat_dirs(float* outb, const float* hfw,
                                                   const float* hbw, const int* lens, int T) {
  size_t i = (size_t)blockIdx.x * 256 + threadIdx.x;   // over B*T*(HD/4)
  size_t total = (size_t)BB * T * (HD / 4);
  if (i >= total) return;
  int j = (int)(i & (HD / 4 - 1)) * 4;
  size_t bt = i >> 5;
  int t = (int)(bt % T);
  int b = (int)(bt / T);
  *(float4*)(outb + bt * NHD + j) = *(const float4*)(hfw + bt * HD + j);
  int len = lens[b];
  int rt = len - 1 - t;
  rt = rt < 0 ? 0 : (rt > T - 1 ? T - 1 : rt);
  float4 v = {0.f, 0.f, 0.f, 0.f};
  if (t < len) v = *(const float4*)(hbw + ((size_t)b * T + rt) * HD + j);
  *(float4*)(outb + bt * NHD + HD + j) = v;
}

// masked softmax over X (rows = B * rpb, b = row / rpb), in place
__global__ __launch_bounds__(256) void masked_softmax(float* s, const int* xlen, int rpb) {
  __shared__ float red[256];
  int rowi = blockIdx.x;
  int b = rowi / rpb;
  int x = threadIdx.x;
  float* rp = s + (size_t)rowi * XDIM;
  float v = (x < xlen[b]) ? rp[x] : NEGV;
  red[x] = v;
  __syncthreads();
  for (int o = 128; o > 0; o >>= 1) {
    if (x < o) red[x] = fmaxf(red[x], red[x + o]);
    __syncthreads();
  }
  float mx = red[0];
  __syncthreads();
  float e = __expf(v - mx);
  red[x] = e;
  __syncthreads();
  for (int o = 128; o > 0; o >>= 1) {
    if (x < o) red[x] += red[x + o];
    __syncthreads();
  }
  rp[x] = e / red[0];
}

// sv[b,x] = sum_c where(mask_both, s[b,c,x], NEG)   (num_head pre-softmax K)
__global__ __launch_bounds__(256) void sv_reduce(const float* s, const int* xlen,
                                                 const int* clen, float* sv) {
  int i = blockIdx.x * 256 + threadIdx.x;
  if (i >= BB * XDIM) return;
  int b = i >> 8, x = i & 255;
  int cl = clen[b];
  bool vx = x < xlen[b];
  const float* sp = s + (size_t)b * CDIM * XDIM + x;
  float acc = 0.f;
  for (int c = 0; c < CDIM; ++c)
    acc += (c < cl && vx) ? sp[(size_t)c * XDIM] : NEGV;
  sv[i] = acc;
}

// K[b,h] = sum_x sv[b,x] * henc[b,x,h]
__global__ __launch_bounds__(256) void gemv_K(const float* sv, const float* henc, float* Kout) {
  int i = blockIdx.x * 256 + threadIdx.x;
  if (i >= BB * NHD) return;
  int b = i >> 8, h = i & 255;
  const float* svb = sv + (size_t)b * XDIM;
  const float* hb  = henc + (size_t)b * XDIM * NHD + h;
  float acc = 0.f;
  for (int x = 0; x < XDIM; ++x) acc += svb[x] * hb[(size_t)x * NHD];
  Kout[i] = acc;
}

// col_emb[b,k,:] = ecol[b, gt_sel[b,k], :]
__global__ __launch_bounds__(256) void gather_colemb(const float* ecol, const int* gt, float* ce) {
  int i = blockIdx.x * 256 + threadIdx.x;
  if (i >= BB * 4 * NHD) return;
  int h = i & 255;
  int bk = i >> 8;
  int k = bk & 3, b = bk >> 2;
  ce[i] = ecol[((size_t)b * CDIM + gt[b * 4 + k]) * NHD + h];
}

__global__ __launch_bounds__(256) void mask_cols_neg(float* sc, const int* clen) {
  int i = blockIdx.x * 256 + threadIdx.x;
  if (i >= BB * CDIM) return;
  int b = i >> 6, c = i & 63;
  if (c >= clen[b]) sc[i] = NEGV;
}

// ===========================================================================
// Host orchestration
// ===========================================================================
extern "C" void kernel_launch(void* const* d_in, const int* in_sizes, int n_in,
                              void* d_out, int out_size, void* d_ws, size_t ws_size,
                              hipStream_t stream) {
  (void)in_sizes; (void)n_in; (void)out_size; (void)ws_size;

  const float* x_emb   = (const float*)d_in[0];
  const float* col_inp = (const float*)d_in[1];
  auto PW = [&](int leaf) { return (const float*)d_in[2 + leaf]; };
  const int* xlen  = (const int*)d_in[2 + N_PARAM_LEAVES + 0];
  const int* clen  = (const int*)d_in[2 + N_PARAM_LEAVES + 1];
  const int* gtsel = (const int*)d_in[2 + N_PARAM_LEAVES + 2];
  float* out = (float*)d_out;

  // --- workspace arena (floats) ---
  float* w = (float*)d_ws;
  size_t off = 0;
  auto A_ = [&](size_t n) { float* p = w + off; off += n; return p; };
  float* ECOL  = A_((size_t)BB * CDIM * NHD);   // col encoder output
  float* HENC  = A_((size_t)BB * XDIM * NHD);   // x encoder output
  float* REVIN = A_((size_t)BB * XDIM * NW);    // reversed LSTM input / SBUF
  float* GX    = A_((size_t)BB * XDIM * 512);   // input gates / UH
  float* HFW   = A_((size_t)BB * XDIM * HD);    // fw hidden / KBUF
  float* HBW   = A_((size_t)BB * XDIM * HD);    // bw hidden / TMP
  float* PING  = A_((size_t)BB * XDIM * NHD);   // layer0 output / small bufs
  float* BIASB = A_(512);
  // phase-2 aliases (encoders finished before heads; stream-ordered)
  float* UH   = GX;
  float* SBUF = REVIN;
  float* KBUF = HFW;
  float* TMP  = HBW;
  float* SV     = PING;
  float* KNUM   = PING + 32768;
  float* COLEMB = PING + 65536;
  float* S4     = COLEMB + 131072;
  float* K4     = S4 + 131072;
  float* T1     = K4 + 131072;
  float* T2     = T1 + 131072;

  hipFuncSetAttribute(reinterpret_cast<const void*>(lstm_recur),
                      hipFuncAttributeMaxDynamicSharedMemorySize, LSTM_SHMEM);

  auto gemm = [&](const float* Am, long long ldA, long long sA,
                  const float* Bm, long long ldB, long long sB, int transB,
                  const float* bias, float* Cm, long long ldC, long long sC,
                  int M, int N, int K, int act, int accum, int batch) {
    dim3 g((unsigned)((N + 127) / 128), (unsigned)((M + 63) / 64), (unsigned)batch);
    gemm_wmma<<<g, dim3(256), 0, stream>>>(Am, ldA, sA, Bm, ldB, sB, transB,
                                           bias, Cm, ldC, sC, M, N, K, act, accum);
  };

  // ---- BiLSTM encoder ----
  auto run_bilstm = [&](const float* seq, int T, const int* lens, int lstm_leaf,
                        float* outenc) {
    const float* layin = seq;
    int din = NW;
    for (int l = 0; l < 2; ++l) {
      float* layout_ = (l == 0) ? PING : outenc;
      for (int dir = 1; dir >= 0; --dir) {  // leaf order: bw=0, fw=1
        int base = lstm_leaf + l * 8 + dir * 4;
        const float* Whh = PW(base + 0);
        const float* Wih = PW(base + 1);
        const float* bhh = PW(base + 2);
        const float* bih = PW(base + 3);
        bias_sum<<<dim3(1), dim3(512), 0, stream>>>(BIASB, bih, bhh);
        const float* gin = layin;
        if (dir == 0) {
          size_t tot = (size_t)BB * T * (din / 4);
          gather_rev<<<dim3((unsigned)((tot + 255) / 256)), dim3(256), 0, stream>>>(
              REVIN, layin, lens, T, din);
          gin = REVIN;
        }
        // Gx[b*T+t, :] = x_t @ Wih^T + (bih+bhh)  (parallel over all t)
        gemm(gin, din, 0, Wih, din, 0, 1, BIASB, GX, 512, 0, BB * T, 512, din, 0, 0, 1);
        float* Hd = dir ? HFW : HBW;
        lstm_recur<<<dim3(BB / 16), dim3(256), LSTM_SHMEM, stream>>>(GX, Whh, lens, Hd, T);
      }
      size_t tot = (size_t)BB * T * (HD / 4);
      concat_dirs<<<dim3((unsigned)((tot + 255) / 256)), dim3(256), 0, stream>>>(
          layout_, HFW, HBW, lens, T);
      layin = layout_;
      din = NHD;
    }
  };

  run_bilstm(col_inp, CDIM, clen, L_lstm_col, ECOL);
  run_bilstm(x_emb, XDIM, xlen, L_lstm_x, HENC);

  gather_colemb<<<dim3((BB * 4 * NHD) / 256), dim3(256), 0, stream>>>(ECOL, gtsel, COLEMB);

  // ---- heads ----
  auto num_head = [&](int Lh, int Ll, int Lout, int outN, float* outp) {
    gemm(HENC, NHD, 0, PW(Lh), NHD, 0, 1, PW(Lh + 1), UH, NHD, 0,
         BB * XDIM, NHD, NHD, 0, 0, 1);                                    // Uh
    gemm(ECOL, NHD, (long long)CDIM * NHD, UH, NHD, (long long)XDIM * NHD, 1,
         nullptr, SBUF, XDIM, (long long)CDIM * XDIM, CDIM, XDIM, NHD, 0, 0, BB); // s
    sv_reduce<<<dim3((BB * XDIM) / 256), dim3(256), 0, stream>>>(SBUF, xlen, clen, SV);
    gemv_K<<<dim3((BB * NHD) / 256), dim3(256), 0, stream>>>(SV, HENC, KNUM);
    gemm(KNUM, NHD, 0, PW(Ll), NHD, 0, 1, PW(Ll + 1), T2, NHD, 0,
         BB, NHD, NHD, 1, 0, 1);                                           // tanh(lin_l)
    gemm(T2, NHD, 0, PW(Lout), NHD, 0, 1, PW(Lout + 1), outp, outN, 0,
         BB, outN, NHD, 0, 0, 1);
  };

  auto col_head = [&](int Latt, int LK, int Lcol, int Lout, float* outp, bool maskc) {
    gemm(HENC, NHD, 0, PW(Latt), NHD, 0, 1, PW(Latt + 1), UH, NHD, 0,
         BB * XDIM, NHD, NHD, 0, 0, 1);
    gemm(ECOL, NHD, (long long)CDIM * NHD, UH, NHD, (long long)XDIM * NHD, 1,
         nullptr, SBUF, XDIM, (long long)CDIM * XDIM, CDIM, XDIM, NHD, 0, 0, BB);
    masked_softmax<<<dim3(BB * CDIM), dim3(256), 0, stream>>>(SBUF, xlen, CDIM);
    gemm(SBUF, XDIM, (long long)CDIM * XDIM, HENC, NHD, (long long)XDIM * NHD, 0,
         nullptr, KBUF, NHD, (long long)CDIM * NHD, CDIM, NHD, XDIM, 0, 0, BB); // K
    gemm(KBUF, NHD, 0, PW(LK), NHD, 0, 1, PW(LK + 1), TMP, NHD, 0,
         BB * CDIM, NHD, NHD, 0, 0, 1);
    gemm(ECOL, NHD, 0, PW(Lcol), NHD, 0, 1, PW(Lcol + 1), TMP, NHD, 0,
         BB * CDIM, NHD, NHD, 1, 1, 1);                                    // tanh(sum)
    gemm(TMP, NHD, 0, PW(Lout), NHD, 0, 1, PW(Lout + 1), outp, 1, 0,
         BB * CDIM, 1, NHD, 0, 0, 1);
    if (maskc)
      mask_cols_neg<<<dim3((BB * CDIM) / 256), dim3(256), 0, stream>>>(outp, clen);
  };

  auto agg_head = [&](int Latt, int LK, int Lcol, int Lh0, int Lh1, int outN, float* outp) {
    gemm(HENC, NHD, 0, PW(Latt), NHD, 0, 1, PW(Latt + 1), UH, NHD, 0,
         BB * XDIM, NHD, NHD, 0, 0, 1);
    gemm(COLEMB, NHD, 4LL * NHD, UH, NHD, (long long)XDIM * NHD, 1, nullptr,
         S4, XDIM, 4LL * XDIM, 4, XDIM, NHD, 0, 0, BB);
    masked_softmax<<<dim3(BB * 4), dim3(256), 0, stream>>>(S4, xlen, 4);
    gemm(S4, XDIM, 4LL * XDIM, HENC, NHD, (long long)XDIM * NHD, 0, nullptr,
         K4, NHD, 4LL * NHD, 4, NHD, XDIM, 0, 0, BB);
    gemm(K4, NHD, 0, PW(LK), NHD, 0, 1, PW(LK + 1), T1, NHD, 0,
         BB * 4, NHD, NHD, 0, 0, 1);
    gemm(COLEMB, NHD, 0, PW(Lcol), NHD, 0, 1, PW(Lcol + 1), T1, NHD, 0,
         BB * 4, NHD, NHD, 0, 1, 1);                                       // sum, no act
    gemm(T1, NHD, 0, PW(Lh0), NHD, 0, 1, PW(Lh0 + 1), T2, NHD, 0,
         BB * 4, NHD, NHD, 1, 0, 1);                                       // tanh
    gemm(T2, NHD, 0, PW(Lh1), NHD, 0, 1, PW(Lh1 + 1), outp, outN, 0,
         BB * 4, outN, NHD, 0, 0, 1);
  };

  // outputs (flat, in tuple order)
  num_head(L_sel_num_h, L_sel_num_l, L_sel_num_out, 4, out + 0);            // [B,4]
  col_head(L_sel_att, L_sel_out_K, L_sel_out_col, L_sel_out, out + 512, true);   // [B,64]
  agg_head(L_agg_num_att, L_agg_num_out_K, L_agg_num_out_col,
           L_agg_num_out_0, L_agg_num_out_1, 4, out + 8704);                // [B,4,4]
  agg_head(L_agg_op_att, L_agg_op_out_K, L_agg_op_out_col,
           L_agg_op_out_0, L_agg_op_out_1, 6, out + 10752);                 // [B,4,6]
  num_head(L_gby_num_h, L_gby_num_l, L_gby_num_out, 3, out + 13824);        // [B,3]
  col_head(L_gby_att, L_gby_out_K, L_gby_out_col, L_gby_out, out + 14208, false); // [B,64]
  num_head(L_ody_num_h, L_ody_num_l, L_ody_num_out, 2, out + 22400);        // [B,2]
  col_head(L_ody_att, L_ody_out_K, L_ody_out_col, L_ody_out, out + 22656, false); // [B,64]
  agg_head(L_ody_agg_att, L_ody_agg_out_K, L_ody_agg_out_col,
           L_ody_agg_out_0, L_ody_agg_out_1, 6, out + 30848);               // [B,4,6]
  num_head(L_ody_par_h, L_ody_par_l, L_ody_par_out, 3, out + 33920);        // [B,3]
}
